// GPT2_44598940401881
// MI455X (gfx1250) — compile-verified, matrix operand order checked by reading, not run on hardware
//
#include <hip/hip_runtime.h>
#include <stdint.h>

typedef __bf16 bf16;
typedef __attribute__((ext_vector_type(16))) __bf16 bf16x16;
typedef __attribute__((ext_vector_type(8)))  __bf16 bf16x8;
typedef __attribute__((ext_vector_type(8)))  float  f32x8;

// ---------- helpers ----------
__device__ __forceinline__ bf16 f2bf(float f) {
    union { float f; unsigned u; } c; c.f = f;
    unsigned u = c.u;
    unsigned r = (u + 0x7FFFu + ((u >> 16) & 1u)) >> 16;  // RNE
    union { unsigned short s; bf16 b; } o; o.s = (unsigned short)r;
    return o.b;
}

__device__ __forceinline__ uint32_t pkbf2(float a, float b) {  // pack 2 bf16 -> dword
    union { bf16 e[2]; uint32_t u; } p;
    p.e[0] = f2bf(a); p.e[1] = f2bf(b);
    return p.u;
}

// Async copy 16B global -> LDS (gfx1250 GLOBAL_LOAD_ASYNC_TO_LDS_B128, ASYNCcnt).
// lds_off: LDS byte offset (low 32 bits of a generic pointer to __shared__).
__device__ __forceinline__ void async_cp16(uint32_t lds_off, const void* gptr) {
    asm volatile("global_load_async_to_lds_b128 %0, %1, off"
                 :: "v"(lds_off), "v"((unsigned long long)(uintptr_t)gptr)
                 : "memory");
}
__device__ __forceinline__ void wait_async0() {
    asm volatile("s_wait_asynccnt 0x0" ::: "memory");
}
__device__ __forceinline__ uint32_t lds_off32(const void* p) {
    return (uint32_t)(uintptr_t)p;   // LDS aperture: low 32 bits = LDS offset
}

// Load a 16x32 bf16 A/B WMMA fragment from K-major LDS (per-lane base p):
// elements [0..7] at p, [8..15] at p+16 (CDNA5 16-bit operand layout).
__device__ __forceinline__ bf16x16 ldfrag(const bf16* p) {
    union U { bf16x16 v; bf16x8 h[2]; } u;
    u.h[0] = *(const bf16x8*)(p);
    u.h[1] = *(const bf16x8*)(p + 16);
    return u.v;
}

__device__ __forceinline__ f32x8 wmma_bf16(bf16x16 a, bf16x16 b, f32x8 c) {
    return __builtin_amdgcn_wmma_f32_16x16x32_bf16(false, a, false, b,
                                                   (short)0, c, false, false);
}

// ---------- embedding ----------
__global__ __launch_bounds__(256) void embed_kernel(
    const int* __restrict__ tok, const float* __restrict__ wte,
    const float* __restrict__ wpe, float* __restrict__ x) {
    int row = blockIdx.x;               // 0..2047 (b*1024 + n)
    int n   = row & 1023;
    int t   = tok[row];
    const float* te = wte + (size_t)t * 768;
    const float* pe = wpe + (size_t)n * 768;
    float* o = x + (size_t)row * 768;
    for (int c = threadIdx.x; c < 768; c += 256) o[c] = te[c] + pe[c];
}

// ---------- layernorm (fp32 in -> bf16 out), one wave32 per row ----------
__global__ __launch_bounds__(256) void ln_kernel(
    const float* __restrict__ x, const float* __restrict__ w,
    const float* __restrict__ b, bf16* __restrict__ out) {
    int wid = threadIdx.x >> 5, lane = threadIdx.x & 31;
    int row = blockIdx.x * 8 + wid;
    const float* xr = x + (size_t)row * 768;
    float v[24];
    float sum = 0.f;
#pragma unroll
    for (int i = 0; i < 24; ++i) { v[i] = xr[lane + i * 32]; sum += v[i]; }
#pragma unroll
    for (int off = 1; off < 32; off <<= 1) sum += __shfl_xor(sum, off, 32);
    float mu = sum * (1.0f / 768.0f);
    float var = 0.f;
#pragma unroll
    for (int i = 0; i < 24; ++i) { float d = v[i] - mu; var += d * d; }
#pragma unroll
    for (int off = 1; off < 32; off <<= 1) var += __shfl_xor(var, off, 32);
    float rs = rsqrtf(var * (1.0f / 768.0f) + 1e-5f);
    bf16* o = out + (size_t)row * 768;
#pragma unroll
    for (int i = 0; i < 24; ++i) {
        int c = lane + i * 32;
        o[c] = f2bf((v[i] - mu) * rs * w[c] + b[c]);
    }
}

// ---------- generic WMMA GEMM (double-buffered, async A staging) ----------
// C[M,N] = epi(A_bf16[M,K] @ W_f32 + bias).  W is [K,N] (BT=false) or [N,K] (BT=true).
// EPI: 0 = store bf16, 1 = residual fp32 +=, 2 = exact GELU -> bf16, 3 = store fp32
#define LDA 40  // LDS K-stride (elements): 80B, 16B aligned, conflict-spreading
template <int EPI, bool BT>
__global__ __launch_bounds__(256) void gemm_kernel(
    const bf16* __restrict__ A, const float* __restrict__ W,
    const float* __restrict__ bias, void* __restrict__ Cout,
    float* __restrict__ resid, int M, int N, int K) {
    __shared__ bf16 As[2 * 128 * LDA];
    __shared__ bf16 Bs[2 * 128 * LDA];

    const int tid  = threadIdx.x;
    const int wid  = tid >> 5, lane = tid & 31;
    const int wm   = wid & 3;           // 4 waves along M: rows wm*32
    const int wn   = wid >> 2;          // 2 waves along N: cols wn*64
    const int hb   = (lane >> 4) & 1;   // half-wave select
    const int l16  = lane & 15;
    const int m0   = blockIdx.y * 128;
    const int n0   = blockIdx.x * 128;
    const int nk   = K >> 5;            // K-steps of 32

    // A staging: each thread async-copies 32B (two b128s) of its 128x32 tile slice
    const int arow = tid >> 1, acol = (tid & 1) * 16;
    auto asyncA = [&](int buf, int k0) {
        const bf16* src = A + (size_t)(m0 + arow) * K + k0 + acol;
        uint32_t dst = lds_off32(&As[buf * 128 * LDA + arow * LDA + acol]);
        async_cp16(dst, src);
        async_cp16(dst + 16, src + 8);
    };
    // B staging: global fp32 -> regs
    auto loadB = [&](int k0, float4* breg) {
        if (!BT) {
#pragma unroll
            for (int it = 0; it < 2; ++it) {          // 512 items: (kpair, n-quad)
                int q  = tid + it * 256;
                int kp = q >> 5, n4 = (q & 31) * 4;
                const float* s0 = W + (size_t)(k0 + 2 * kp) * N + n0 + n4;
                breg[it * 2 + 0] = *(const float4*)(s0);
                breg[it * 2 + 1] = *(const float4*)(s0 + N);
            }
        } else {
#pragma unroll
            for (int it = 0; it < 4; ++it) {          // 1024 items: (n, k-quad)
                int q  = tid + it * 256;
                int nn = q >> 3, k4 = (q & 7) * 4;
                float4 f = make_float4(0.f, 0.f, 0.f, 0.f);
                if (n0 + nn < N)
                    f = *(const float4*)(W + (size_t)(n0 + nn) * K + k0 + k4);
                breg[it] = f;
            }
        }
    };
    // regs -> LDS (bf16, K-major Bt[n][k]) with packed dword stores
    auto storeB = [&](int buf, const float4* breg) {
        bf16* Bb = &Bs[buf * 128 * LDA];
        if (!BT) {
#pragma unroll
            for (int it = 0; it < 2; ++it) {
                int q  = tid + it * 256;
                int kp = q >> 5, n4 = (q & 31) * 4;
                const float4 f0 = breg[it * 2 + 0], f1 = breg[it * 2 + 1];
                *(uint32_t*)&Bb[(n4 + 0) * LDA + 2 * kp] = pkbf2(f0.x, f1.x);
                *(uint32_t*)&Bb[(n4 + 1) * LDA + 2 * kp] = pkbf2(f0.y, f1.y);
                *(uint32_t*)&Bb[(n4 + 2) * LDA + 2 * kp] = pkbf2(f0.z, f1.z);
                *(uint32_t*)&Bb[(n4 + 3) * LDA + 2 * kp] = pkbf2(f0.w, f1.w);
            }
        } else {
#pragma unroll
            for (int it = 0; it < 4; ++it) {
                int q  = tid + it * 256;
                int nn = q >> 3, k4 = (q & 7) * 4;
                const float4 f = breg[it];
                uint2 s; s.x = pkbf2(f.x, f.y); s.y = pkbf2(f.z, f.w);
                *(uint2*)&Bb[nn * LDA + k4] = s;
            }
        }
    };

    f32x8 acc[2][4] = {};
    float4 breg[4];

    // prologue: stage tile 0
    asyncA(0, 0);
    loadB(0, breg);
    storeB(0, breg);
    wait_async0();
    __syncthreads();

    for (int kt = 0; kt < nk; ++kt) {
        const int cur = kt & 1, nxt = cur ^ 1;
        // kick off next tile's global traffic before computing on current
        if (kt + 1 < nk) {
            asyncA(nxt, (kt + 1) * 32);
            loadB((kt + 1) * 32, breg);
        }

        const bf16* Ab = &As[cur * 128 * LDA];
        const bf16* Bb = &Bs[cur * 128 * LDA];
        bf16x16 af[2], bfr[4];
#pragma unroll
        for (int i = 0; i < 2; ++i)
            af[i] = ldfrag(&Ab[(wm * 32 + i * 16 + l16) * LDA + hb * 8]);
#pragma unroll
        for (int j = 0; j < 4; ++j)
            bfr[j] = ldfrag(&Bb[(wn * 64 + j * 16 + l16) * LDA + hb * 8]);
#pragma unroll
        for (int i = 0; i < 2; ++i)
#pragma unroll
            for (int j = 0; j < 4; ++j)
                acc[i][j] = wmma_bf16(af[i], bfr[j], acc[i][j]);

        if (kt + 1 < nk) storeB(nxt, breg);
        wait_async0();
        __syncthreads();
    }

    // ---- epilogue (C layout: VGPR r -> row r + 8*hb, lane%16 -> col)
#pragma unroll
    for (int i = 0; i < 2; ++i)
#pragma unroll
        for (int j = 0; j < 4; ++j)
#pragma unroll
            for (int r = 0; r < 8; ++r) {
                int row = m0 + wm * 32 + i * 16 + r + hb * 8;
                int col = n0 + wn * 64 + j * 16 + l16;
                if (col < N) {
                    float v = acc[i][j][r] + (bias ? bias[col] : 0.f);
                    if (EPI == 0) {
                        ((bf16*)Cout)[(size_t)row * N + col] = f2bf(v);
                    } else if (EPI == 1) {
                        resid[(size_t)row * N + col] += v;
                    } else if (EPI == 2) {
                        float g = 0.5f * v * (1.0f + erff(v * 0.7071067811865476f));
                        ((bf16*)Cout)[(size_t)row * N + col] = f2bf(g);
                    } else {
                        ((float*)Cout)[(size_t)row * N + col] = v;
                    }
                }
            }
}

// ---------- flash attention (causal), one block per (b, h, 128-row q tile) ----------
__global__ __launch_bounds__(256) void attn_kernel(
    const bf16* __restrict__ qkv, bf16* __restrict__ out) {
    __shared__ bf16 Qs[128 * 72];       // [q_row][d]
    __shared__ bf16 Ks[128 * 72];       // [key][d]
    __shared__ bf16 Vt[64 * 136];       // [d][key] (transposed for B operand)
    __shared__ bf16 Ps[8 * 16 * 136];   // per-wave P tile [row][key]

    const int qt  = blockIdx.x & 7;
    const int bh  = blockIdx.x >> 3;
    const int b   = bh / 12, h = bh % 12;
    const int tid = threadIdx.x;
    const int wid = tid >> 5, lane = tid & 31;
    const int hb  = (lane >> 4) & 1, l16 = lane & 15;
    const int r0  = wid * 16;           // wave's 16 query rows within the tile

    const bf16* qbase = qkv + (size_t)(b * 1024) * 2304 + h * 64;
    const bf16* kbase = qbase + 768;
    const bf16* vbase = qbase + 1536;

    // stage Q tile (128 x 64 bf16) via async copies: 32 elems / thread
    {
        int row = tid >> 1, seg = (tid & 1) * 32;
        const bf16* src = qbase + (size_t)(qt * 128 + row) * 2304 + seg;
        uint32_t dst = lds_off32(&Qs[row * 72 + seg]);
#pragma unroll
        for (int c = 0; c < 4; ++c) async_cp16(dst + c * 16, src + c * 8);
    }

    float mrow[8], lrow[8];
    f32x8 Ov[4] = {};
#pragma unroll
    for (int r = 0; r < 8; ++r) { mrow[r] = -1e30f; lrow[r] = 0.f; }

    for (int kt = 0; kt <= qt; ++kt) {
        __syncthreads();
        // stage K tile [key][d] (async)
        {
            int row = tid >> 1, seg = (tid & 1) * 32;
            const bf16* src = kbase + (size_t)(kt * 128 + row) * 2304 + seg;
            uint32_t dst = lds_off32(&Ks[row * 72 + seg]);
#pragma unroll
            for (int c = 0; c < 4; ++c) async_cp16(dst + c * 16, src + c * 8);
        }
        // stage V transposed: Vt[d][key], packed dword stores over key pairs
#pragma unroll
        for (int it = 0; it < 4; ++it) {
            int q = tid + it * 256;            // 1024 items: (keypair, d-quad)
            int kp = q >> 4, d4 = (q & 15) * 4;
            union { uint2 u; bf16 e[4]; } a, c;
            const bf16* src = vbase + (size_t)(kt * 128 + 2 * kp) * 2304 + d4;
            a.u = *(const uint2*)(src);
            c.u = *(const uint2*)(src + 2304);
#pragma unroll
            for (int i = 0; i < 4; ++i) {
                union { bf16 e[2]; uint32_t u; } p;
                p.e[0] = a.e[i]; p.e[1] = c.e[i];
                *(uint32_t*)&Vt[(d4 + i) * 136 + 2 * kp] = p.u;
            }
        }
        wait_async0();
        __syncthreads();

        // S = Q K^T  (wave: 16 q rows x 128 keys = 8 C tiles)
        f32x8 sacc[8] = {};
#pragma unroll
        for (int dk = 0; dk < 64; dk += 32) {
            bf16x16 aq = ldfrag(&Qs[(r0 + l16) * 72 + dk + hb * 8]);
#pragma unroll
            for (int j = 0; j < 8; ++j) {
                bf16x16 bk = ldfrag(&Ks[(j * 16 + l16) * 72 + dk + hb * 8]);
                sacc[j] = wmma_bf16(aq, bk, sacc[j]);
            }
        }

        // scale + causal mask (diagonal tile only)
        const int qrow_base = qt * 128 + r0 + hb * 8;
        const int kcol = kt * 128 + l16;
#pragma unroll
        for (int j = 0; j < 8; ++j)
#pragma unroll
            for (int r = 0; r < 8; ++r) {
                float s = sacc[j][r] * 0.125f;
                if (kt == qt && (kcol + j * 16) > (qrow_base + r)) s = -1e30f;
                sacc[j][r] = s;
            }

        // online softmax (rows live in half-wave: 16 lanes per row group)
        float mnew[8], ladd[8];
#pragma unroll
        for (int r = 0; r < 8; ++r) {
            float mx = sacc[0][r];
#pragma unroll
            for (int j = 1; j < 8; ++j) mx = fmaxf(mx, sacc[j][r]);
#pragma unroll
            for (int off = 1; off < 16; off <<= 1)
                mx = fmaxf(mx, __shfl_xor(mx, off, 32));
            mnew[r] = fmaxf(mrow[r], mx);
        }
#pragma unroll
        for (int r = 0; r < 8; ++r) {
            float sum = 0.f;
#pragma unroll
            for (int j = 0; j < 8; ++j) {
                float p = __expf(sacc[j][r] - mnew[r]);
                sacc[j][r] = p;
                sum += p;
            }
#pragma unroll
            for (int off = 1; off < 16; off <<= 1)
                sum += __shfl_xor(sum, off, 32);
            ladd[r] = sum;
        }
#pragma unroll
        for (int r = 0; r < 8; ++r) {
            float alpha = __expf(mrow[r] - mnew[r]);
            lrow[r] = lrow[r] * alpha + ladd[r];
            mrow[r] = mnew[r];
#pragma unroll
            for (int j4 = 0; j4 < 4; ++j4) Ov[j4][r] *= alpha;
        }

        // C-layout -> A-layout via per-wave LDS round trip (DS ops in-order per wave)
        bf16* P = &Ps[wid * 16 * 136];
#pragma unroll
        for (int j = 0; j < 8; ++j)
#pragma unroll
            for (int r = 0; r < 8; ++r)
                P[(r + hb * 8) * 136 + j * 16 + l16] = f2bf(sacc[j][r]);

        // O += P @ V  (K = 128 keys, 4 k-steps)
#pragma unroll
        for (int kk = 0; kk < 128; kk += 32) {
            bf16x16 ap = ldfrag(&P[l16 * 136 + kk + hb * 8]);
#pragma unroll
            for (int j4 = 0; j4 < 4; ++j4) {
                bf16x16 bv = ldfrag(&Vt[(j4 * 16 + l16) * 136 + kk + hb * 8]);
                Ov[j4] = wmma_bf16(ap, bv, Ov[j4]);
            }
        }
    }

    // write attn output bf16: [b*1024+n, h*64+d]
#pragma unroll
    for (int j4 = 0; j4 < 4; ++j4)
#pragma unroll
        for (int r = 0; r < 8; ++r) {
            int row = b * 1024 + qt * 128 + r0 + r + hb * 8;
            int col = h * 64 + j4 * 16 + l16;
            out[(size_t)row * 768 + col] = f2bf(Ov[j4][r] / lrow[r]);
        }
}

// ---------- host orchestration ----------
extern "C" void kernel_launch(void* const* d_in, const int* in_sizes, int n_in,
                              void* d_out, int out_size, void* d_ws, size_t ws_size,
                              hipStream_t stream) {
    (void)in_sizes; (void)n_in; (void)out_size; (void)ws_size;
    const int*   tok    = (const int*)  d_in[0];
    const float* wte    = (const float*)d_in[1];
    const float* wpe    = (const float*)d_in[2];
    const float* ln1w   = (const float*)d_in[3];
    const float* ln1b   = (const float*)d_in[4];
    const float* attnw  = (const float*)d_in[5];
    const float* attnb  = (const float*)d_in[6];
    const float* aprojw = (const float*)d_in[7];
    const float* aprojb = (const float*)d_in[8];
    const float* ln2w   = (const float*)d_in[9];
    const float* ln2b   = (const float*)d_in[10];
    const float* fcw    = (const float*)d_in[11];
    const float* fcb    = (const float*)d_in[12];
    const float* mprojw = (const float*)d_in[13];
    const float* mprojb = (const float*)d_in[14];
    const float* lnfw   = (const float*)d_in[15];
    const float* lnfb   = (const float*)d_in[16];

    char* ws = (char*)d_ws;
    float* X   = (float*)(ws);                         //  6,291,456 B  fp32 residual [2048,768]
    bf16*  Hb  = (bf16*) (ws + 6291456);               //  3,145,728 B  ln output bf16
    bf16*  QKV = (bf16*) (ws + 9437184);               //  9,437,184 B  qkv bf16 [2048,2304]
    bf16*  AT  = (bf16*) (ws + 18874368);              //  3,145,728 B  attn out bf16
    bf16*  FF  = (bf16*) (ws + 22020096);              // 12,582,912 B  gelu(fc) bf16 [2048,3072]

    const int M = 2048;

    embed_kernel<<<M, 256, 0, stream>>>(tok, wte, wpe, X);

    for (int l = 0; l < 12; ++l) {
        ln_kernel<<<M / 8, 256, 0, stream>>>(X, ln1w + l * 768, ln1b + l * 768, Hb);
        gemm_kernel<0, false><<<dim3(18, 16), 256, 0, stream>>>(
            Hb, attnw + (size_t)l * 768 * 2304, attnb + (size_t)l * 2304,
            (void*)QKV, nullptr, M, 2304, 768);
        attn_kernel<<<192, 256, 0, stream>>>(QKV, AT);
        gemm_kernel<1, false><<<dim3(6, 16), 256, 0, stream>>>(
            AT, aprojw + (size_t)l * 768 * 768, aprojb + (size_t)l * 768,
            nullptr, X, M, 768, 768);
        ln_kernel<<<M / 8, 256, 0, stream>>>(X, ln2w + l * 768, ln2b + l * 768, Hb);
        gemm_kernel<2, false><<<dim3(24, 16), 256, 0, stream>>>(
            Hb, fcw + (size_t)l * 768 * 3072, fcb + (size_t)l * 3072,
            (void*)FF, nullptr, M, 3072, 768);
        gemm_kernel<1, false><<<dim3(6, 16), 256, 0, stream>>>(
            FF, mprojw + (size_t)l * 3072 * 768, mprojb + (size_t)l * 768,
            nullptr, X, M, 768, 3072);
    }

    ln_kernel<<<M / 8, 256, 0, stream>>>(X, lnfw, lnfb, Hb);
    // tied lm_head: logits = Hb @ wte^T   (wte is [V, D], BT path, no bias)
    gemm_kernel<3, true><<<dim3((50257 + 127) / 128, 16), 256, 0, stream>>>(
        Hb, wte, nullptr, d_out, nullptr, M, 50257, 768);
}
